// SCARM_82325933130021
// MI455X (gfx1250) — matrix operand order repaired and checked by reading, not compiled, not run on hardware
//
#include <hip/hip_runtime.h>
#include <hip/hip_bf16.h>
#include <math.h>

typedef __bf16 bf16_t;
typedef __bf16 v16bf __attribute__((ext_vector_type(16)));
typedef __bf16 v8bf  __attribute__((ext_vector_type(8)));
typedef __bf16 v4bf  __attribute__((ext_vector_type(4)));
typedef float  v8f   __attribute__((ext_vector_type(8)));

#define IMG 16384   // 128*128
#define NF  65
#define HF  8320    // 128*65

// Fragment layouts (CDNA5 ISA):
//  A 16x32 bf16 : lane l, row m=l&15 ; elements 0..7 -> K = k0+8g+e (contig), 8..15 -> K = k0+16+8g+(e-8) (contig), g=l>>4
//  B 32x16 bf16 : lane l, col n=l&15 ; same K mapping as A (tile stored n-major), h=l>>4
//  C/D f32      : lane l, col n=l&15 ; reg r -> m = 8h + r

__device__ __forceinline__ v16bf frag_cat(v8bf lo, v8bf hi) {
  return __builtin_shufflevector(lo, hi, 0, 1, 2, 3, 4, 5, 6, 7, 8, 9, 10, 11, 12, 13, 14, 15);
}

// ---------------- Generic batched WMMA GEMM ----------------
// C[m,n] = op( alpha * sum_k A[m,k]*B(k,n) * scale[m] + shift[m] )
// A must have >=64 rows and Kp columns (zero padded). op: 0=none 1=clip 2=hardswish 3=leaky
__global__ __launch_bounds__(128) void wmma_gemm_kernel(
    const bf16_t* __restrict__ A, long a_so, long a_si, int lda,
    const bf16_t* __restrict__ B, long b_so, long b_si, int ldb, int btrans,
    float* __restrict__ C, bf16_t* __restrict__ Cb, long c_so, long c_si, int ldc,
    int binner, int M, int N, int K, int Kp,
    float alpha, const float* __restrict__ scale, const float* __restrict__ shift,
    int op, float slope)
{
  __shared__ __align__(16) bf16_t Bs[16 * 32];   // n-major [n][k]
  int batch = blockIdx.y;
  int bo = batch / binner, bi = batch % binner;
  const bf16_t* Ab = A + (long)bo * a_so + (long)bi * a_si;
  const bf16_t* Bb = B + (long)bo * b_so + (long)bi * b_si;
  long coff = (long)bo * c_so + (long)bi * c_si;
  int n0 = blockIdx.x * 16;
  int l = threadIdx.x & 31;
  int m_base = (threadIdx.x >> 5) * 16;
  int m = m_base + (l & 15);
  int g8 = (l >> 4) << 3;         // 8g
  const bf16_t* bp = Bs + (l & 15) * 32 + g8;  // n*32 + 8h
  v8f acc = {};
  int nk = Kp >> 5;
  for (int kc = 0; kc < nk; ++kc) {
    int k0 = kc << 5;
    if (btrans) {                  // full-K path (K==Kp): pure vector copy
      int n = threadIdx.x >> 3, kq = (threadIdx.x & 7) << 2;
      *(v4bf*)(Bs + n * 32 + kq) = *(const v4bf*)(Bb + (long)(n0 + n) * ldb + k0 + kq);
    } else {                       // [k][n] source: transpose while staging
      int k = threadIdx.x >> 2, j = (threadIdx.x & 3) << 2;
      v4bf v = {};
      int gk = k0 + k;
      if (gk < K) v = *(const v4bf*)(Bb + (long)gk * ldb + n0 + j);
      Bs[(j + 0) * 32 + k] = v[0];
      Bs[(j + 1) * 32 + k] = v[1];
      Bs[(j + 2) * 32 + k] = v[2];
      Bs[(j + 3) * 32 + k] = v[3];
    }
    __syncthreads();
    const bf16_t* ap = Ab + (long)m * lda + k0 + g8;
    v16bf af  = frag_cat(*(const v8bf*)ap, *(const v8bf*)(ap + 16));
    v16bf bfr = frag_cat(*(const v8bf*)bp, *(const v8bf*)(bp + 16));
    acc = __builtin_amdgcn_wmma_f32_16x16x32_bf16(false, af, false, bfr, (short)0, acc, false, false);
    __syncthreads();
  }
#pragma unroll
  for (int r = 0; r < 8; ++r) {
    int mm = m_base + ((l >> 4) << 3) + r;
    int n = n0 + (l & 15);
    if (mm < M) {
      float v = acc[r] * alpha;
      if (scale) v *= scale[mm];
      if (shift) v += shift[mm];
      if (op == 1) v = fminf(fmaxf(v, -1.f), 1.f);
      else if (op == 2) v = v * fminf(fmaxf(v + 3.f, 0.f), 6.f) * (1.f / 6.f);
      else if (op == 3) v = (v >= 0.f) ? v : v * slope;
      long o = coff + (long)mm * ldc + n;
      if (C)  C[o]  = v;
      if (Cb) Cb[o] = (bf16_t)v;
    }
  }
}

// ---------------- Implicit-GEMM 3x3 conv (Cout=64), dilation d, pad d ----------------
// Wt layout: [9][64][Cin] bf16  (tap-major, contiguous ci)
__global__ __launch_bounds__(128) void wmma_conv3x3_kernel(
    const bf16_t* __restrict__ X, long x_img,
    const bf16_t* __restrict__ Wt,
    const float* __restrict__ bias,
    const float* __restrict__ res, long r_img,
    float* __restrict__ Y, bf16_t* __restrict__ Yb, long y_img,
    int Cin, int dil, int do_leaky, float slope)
{
  __shared__ __align__(16) bf16_t Bs[16 * 32];   // n-major [n][k]
  int img = blockIdx.y;
  int p0 = blockIdx.x * 16;
  int yy = p0 >> 7;
  int x0 = p0 & 127;
  const bf16_t* Xb = X + (long)img * x_img;
  int l = threadIdx.x & 31;
  int co_base = (threadIdx.x >> 5) * 16;
  int mrow = co_base + (l & 15);
  int g8 = (l >> 4) << 3;
  const bf16_t* bp = Bs + (l & 15) * 32 + g8;
  int ks = threadIdx.x >> 2;            // staging k (ci within chunk)
  int js = (threadIdx.x & 3) << 2;      // staging n group
  v8f acc = {};
  int nkc = Cin >> 5;
  for (int tap = 0; tap < 9; ++tap) {
    int ky = tap / 3 - 1, kx = tap % 3 - 1;
    int iy = yy + dil * ky;
    bool yok = (iy >= 0 && iy < 128);
    int xoff = x0 + dil * kx;
    for (int kc = 0; kc < nkc; ++kc) {
      int c0 = kc << 5;
      const bf16_t* rowp = Xb + ((long)(c0 + ks) << 14) + (iy << 7);
#pragma unroll
      for (int t = 0; t < 4; ++t) {
        int n = js + t;
        int ix = xoff + n;
        bf16_t v = (bf16_t)0.f;
        if (yok && ix >= 0 && ix < 128) v = rowp[ix];
        Bs[n * 32 + ks] = v;
      }
      __syncthreads();
      const bf16_t* ap = Wt + ((long)tap * 64 + mrow) * Cin + c0 + g8;
      v16bf af  = frag_cat(*(const v8bf*)ap, *(const v8bf*)(ap + 16));
      v16bf bfr = frag_cat(*(const v8bf*)bp, *(const v8bf*)(bp + 16));
      acc = __builtin_amdgcn_wmma_f32_16x16x32_bf16(false, af, false, bfr, (short)0, acc, false, false);
      __syncthreads();
    }
  }
#pragma unroll
  for (int r = 0; r < 8; ++r) {
    int co = co_base + ((l >> 4) << 3) + r;
    int n = l & 15;
    long sp = ((long)co << 14) + (yy << 7) + x0 + n;
    float v = acc[r];
    if (bias) v += bias[co];
    if (do_leaky) v = (v >= 0.f) ? v : v * slope;
    if (res) v += res[(long)img * r_img + sp];
    long o = (long)img * y_img + sp;
    if (Y)  Y[o]  = v;
    if (Yb) Yb[o] = (bf16_t)v;
  }
}

// ---------------- Depthwise 3x3 (pad 1), f32 in, f32/bf16 out ----------------
__global__ void dw3x3_kernel(const float* __restrict__ X, const float* __restrict__ Wd,
                             const float* __restrict__ bias,
                             float* __restrict__ Yf, bf16_t* __restrict__ Yb,
                             int C, long total)
{
  for (long idx = blockIdx.x * (long)blockDim.x + threadIdx.x; idx < total;
       idx += (long)gridDim.x * blockDim.x) {
    int xx = (int)(idx & 127);
    int yy = (int)((idx >> 7) & 127);
    long bc = idx >> 14;
    int c = (int)(bc % C);
    const float* base = X + (bc << 14);
    float s = 0.f;
#pragma unroll
    for (int t = 0; t < 9; ++t) {
      int iy = yy + t / 3 - 1, ix = xx + t % 3 - 1;
      if (iy >= 0 && iy < 128 && ix >= 0 && ix < 128)
        s += base[(iy << 7) + ix] * Wd[c * 9 + t];
    }
    if (bias) s += bias[c];
    if (Yf) Yf[idx] = s;
    if (Yb) Yb[idx] = (bf16_t)s;
  }
}

// ---------------- Deformable depthwise 3x3 (bilinear) ----------------
__global__ void deform_kernel(const float* __restrict__ X, const float* __restrict__ OFF,
                              const float* __restrict__ Wd,
                              float* __restrict__ Yf, bf16_t* __restrict__ Yb,
                              int C, long total)
{
  for (long idx = blockIdx.x * (long)blockDim.x + threadIdx.x; idx < total;
       idx += (long)gridDim.x * blockDim.x) {
    int xx = (int)(idx & 127);
    int yy = (int)((idx >> 7) & 127);
    long bc = idx >> 14;
    int c = (int)(bc % C);
    long b = bc / C;
    const float* img  = X + (bc << 14);
    const float* offb = OFF + b * 18 * (long)IMG + (yy << 7) + xx;
    float acc = 0.f;
    for (int k = 0; k < 9; ++k) {
      float oy = offb[(long)(2 * k) * IMG];
      float ox = offb[(long)(2 * k + 1) * IMG];
      float py = (float)yy + (float)(k / 3 - 1) + oy;
      float px = (float)xx + (float)(k % 3 - 1) + ox;
      float y0 = floorf(py), x0 = floorf(px);
      float wy = py - y0, wx = px - x0;
      int y0i = (int)y0, x0i = (int)x0;
      float val = 0.f;
#pragma unroll
      for (int d = 0; d < 4; ++d) {
        int dy = d >> 1, dx = d & 1;
        int iy = y0i + dy, ix = x0i + dx;
        float w = (dy ? wy : 1.f - wy) * (dx ? wx : 1.f - wx);
        if (iy >= 0 && iy < 128 && ix >= 0 && ix < 128)
          val += img[(iy << 7) + ix] * w;
      }
      acc += val * Wd[c * 9 + k];
    }
    if (Yf) Yf[idx] = acc;
    if (Yb) Yb[idx] = (bf16_t)acc;
  }
}

// ---------------- DFT kernels (separable rfft2 / irfft2) ----------------
__global__ void dft_row_kernel(const float* __restrict__ X, float* __restrict__ RE, float* __restrict__ IM)
{
  __shared__ float row[128];
  long line = blockIdx.x;
  int t = threadIdx.x;
  row[t] = X[line * 128 + t];
  __syncthreads();
  if (t < NF) {
    float wstep = -6.283185307179586f * (float)t / 128.f;
    float sr = 0.f, si = 0.f;
    for (int n = 0; n < 128; ++n) {
      float s, c;
      __sincosf(wstep * (float)n, &s, &c);
      sr += row[n] * c;
      si += row[n] * s;
    }
    RE[line * NF + t] = sr;
    IM[line * NF + t] = si;
  }
}

__global__ void dft_col_kernel(const float* __restrict__ RE, const float* __restrict__ IM,
                               float* __restrict__ ORE, float* __restrict__ OIM,
                               float sign, float scl)
{
  __shared__ float sre[128], sim[128];
  long img = blockIdx.x / NF;
  int kx = (int)(blockIdx.x % NF);
  int t = threadIdx.x;
  long base = img * (long)HF + kx;
  sre[t] = RE[base + (long)t * NF];
  sim[t] = IM[base + (long)t * NF];
  __syncthreads();
  float wstep = sign * 6.283185307179586f * (float)t / 128.f;
  float sr = 0.f, si = 0.f;
  for (int y = 0; y < 128; ++y) {
    float s, c;
    __sincosf(wstep * (float)y, &s, &c);
    sr += sre[y] * c - sim[y] * s;
    si += sre[y] * s + sim[y] * c;
  }
  ORE[base + (long)t * NF] = sr * scl;
  OIM[base + (long)t * NF] = si * scl;
}

__global__ void idft_row_add_kernel(const float* __restrict__ RE, const float* __restrict__ IM,
                                    const float* __restrict__ H, float* __restrict__ OUT)
{
  __shared__ float sre[NF], sim[NF];
  long line = blockIdx.x;
  int t = threadIdx.x;
  if (t < NF) {
    sre[t] = RE[line * NF + t];
    sim[t] = IM[line * NF + t];
  }
  __syncthreads();
  float s = 0.f;
  for (int k = 0; k < NF; ++k) {
    float w = (k == 0 || k == 64) ? 1.f : 2.f;
    float sn, cs;
    __sincosf(6.283185307179586f * (float)k * (float)t / 128.f, &sn, &cs);
    s += w * (sre[k] * cs - sim[k] * sn);
  }
  OUT[line * 128 + t] = s * (1.f / 128.f) + H[line * 128 + t];
}

__global__ void magpha_kernel(const float* __restrict__ RE, const float* __restrict__ IM,
                              float* __restrict__ MAG, float* __restrict__ PHA, long n)
{
  for (long i = blockIdx.x * (long)blockDim.x + threadIdx.x; i < n;
       i += (long)gridDim.x * blockDim.x) {
    float re = RE[i], im = IM[i];
    MAG[i] = sqrtf(re * re + im * im);
    PHA[i] = atan2f(im, re);
  }
}

__global__ void polar_kernel(const float* __restrict__ MAG, const float* __restrict__ PHA,
                             float* __restrict__ RE, float* __restrict__ IM, long n)
{
  for (long i = blockIdx.x * (long)blockDim.x + threadIdx.x; i < n;
       i += (long)gridDim.x * blockDim.x) {
    float s, c;
    __sincosf(PHA[i], &s, &c);
    RE[i] = MAG[i] * c;
    IM[i] = MAG[i] * s;
  }
}

// ---------------- softmax over rows of 64, bf16 probs out ----------------
__global__ void softmax64_kernel(const float* __restrict__ S, bf16_t* __restrict__ P, int nrows)
{
  int row = blockIdx.x * blockDim.x + threadIdx.x;
  if (row >= nrows) return;
  const float* s = S + (long)row * 64;
  bf16_t* p = P + (long)row * 64;
  float mx = s[0];
  for (int i = 1; i < 64; ++i) mx = fmaxf(mx, s[i]);
  float sum = 0.f;
  for (int i = 0; i < 64; ++i) sum += __expf(s[i] - mx);
  float inv = 1.f / sum;
  for (int i = 0; i < 64; ++i) p[i] = (bf16_t)(__expf(s[i] - mx) * inv);
}

// ---------------- small direct conv: 6->3 3x3 pad1 + bias + residual ----------------
__global__ void final_conv_kernel(const float* __restrict__ A, const float* __restrict__ Bv,
                                  const float* __restrict__ W, const float* __restrict__ bias,
                                  const float* __restrict__ resid, float* __restrict__ out,
                                  long total)
{
  for (long idx = blockIdx.x * (long)blockDim.x + threadIdx.x; idx < total;
       idx += (long)gridDim.x * blockDim.x) {
    int xx = (int)(idx & 127);
    int yy = (int)((idx >> 7) & 127);
    long bco = idx >> 14;
    int co = (int)(bco % 3);
    long b = bco / 3;
    float s = bias[co];
    for (int ci = 0; ci < 6; ++ci) {
      const float* src = (ci < 3) ? (A + ((b * 3 + ci) << 14)) : (Bv + ((b * 3 + ci - 3) << 14));
      const float* wp = W + (co * 6 + ci) * 9;
#pragma unroll
      for (int t = 0; t < 9; ++t) {
        int iy = yy + t / 3 - 1, ix = xx + t % 3 - 1;
        if (iy >= 0 && iy < 128 && ix >= 0 && ix < 128)
          s += src[(iy << 7) + ix] * wp[t];
      }
    }
    out[idx] = s + resid[idx];
  }
}

// ---------------- copy / convert / weight prep ----------------
__global__ void cvt_kernel(const float* __restrict__ src, long s_img,
                           float* __restrict__ dstf, bf16_t* __restrict__ dstb, long d_img,
                           long per, int nimg)
{
  long total = per * nimg;
  for (long i = blockIdx.x * (long)blockDim.x + threadIdx.x; i < total;
       i += (long)gridDim.x * blockDim.x) {
    long img = i / per, r = i % per;
    float v = src[img * s_img + r];
    if (dstf) dstf[img * d_img + r] = v;
    if (dstb) dstb[img * d_img + r] = (bf16_t)v;
  }
}

__global__ void copyb_kernel(const bf16_t* __restrict__ src, long s_img,
                             bf16_t* __restrict__ dst, long d_img, long per, int nimg)
{
  long total = per * nimg;
  for (long i = blockIdx.x * (long)blockDim.x + threadIdx.x; i < total;
       i += (long)gridDim.x * blockDim.x) {
    long img = i / per, r = i % per;
    dst[img * d_img + r] = src[img * s_img + r];
  }
}

// pad [M][K] f32 -> [64][Kp] bf16 (zero padded)
__global__ void padw_kernel(const float* __restrict__ src, bf16_t* __restrict__ dst,
                            int M, int K, int Kp, long total)
{
  for (long i = blockIdx.x * (long)blockDim.x + threadIdx.x; i < total;
       i += (long)gridDim.x * blockDim.x) {
    int m = (int)(i / Kp), k = (int)(i % Kp);
    dst[i] = (m < M && k < K) ? (bf16_t)src[(long)m * K + k] : (bf16_t)0.f;
  }
}

// repack conv weights [L][64][Ci][9] f32 -> [L][9][64][Ci] bf16
__global__ void packconvw_kernel(const float* __restrict__ src, bf16_t* __restrict__ dst,
                                 int Ci, long total)
{
  long per = 9L * 64 * Ci;
  for (long i = blockIdx.x * (long)blockDim.x + threadIdx.x; i < total;
       i += (long)gridDim.x * blockDim.x) {
    long layer = i / per;
    long r = i % per;
    int tap = (int)(r / (64L * Ci));
    int rr = (int)(r % (64L * Ci));
    int co = rr / Ci, ci = rr % Ci;
    dst[i] = (bf16_t)src[((layer * 64 + co) * (long)Ci + ci) * 9 + tap];
  }
}

// ==================================================================================
extern "C" void kernel_launch(void* const* d_in, const int* in_sizes, int n_in,
                              void* d_out, int out_size, void* d_ws, size_t ws_size,
                              hipStream_t stream) {
  (void)in_sizes; (void)n_in; (void)out_size; (void)ws_size;

  const float* x         = (const float*)d_in[0];
  const float* ch_off_dw = (const float*)d_in[1];
  const float* ch_off_pw = (const float*)d_in[2];
  const float* ch_dcn_w  = (const float*)d_in[3];
  const float* ch_pw     = (const float*)d_in[4];
  const float* ct_off_dw = (const float*)d_in[5];
  const float* ct_off_pw = (const float*)d_in[6];
  const float* ct_dcn_w  = (const float*)d_in[7];
  const float* ct_pw     = (const float*)d_in[8];
  const float* db_w      = (const float*)d_in[9];
  const float* db_b      = (const float*)d_in[10];
  const float* ca_dw_w   = (const float*)d_in[11];
  const float* ca_dw_b   = (const float*)d_in[12];
  const float* ca_pw_w   = (const float*)d_in[13];
  const float* ca_pw_b   = (const float*)d_in[14];
  const float* convHH_w  = (const float*)d_in[15];
  const float* convHH_b  = (const float*)d_in[16];
  const float* fd1_w     = (const float*)d_in[17];
  const float* fd2_w     = (const float*)d_in[18];
  const float* fd1_g     = (const float*)d_in[19];
  const float* fd1_bt    = (const float*)d_in[20];
  const float* fd2_g     = (const float*)d_in[21];
  const float* fd2_bt    = (const float*)d_in[22];
  const float* fconv_w   = (const float*)d_in[23];
  const float* conv_w    = (const float*)d_in[24];
  const float* conv_b    = (const float*)d_in[25];

  float* out_ptr = (float*)d_out;
  float* mag_ptr = out_ptr + 12L * 3 * IMG;
  float* pha_ptr = mag_ptr + 12L * 64 * HF;

  char* wp = (char*)d_ws;
  auto alloc = [&](size_t bytes) -> void* {
    void* p = (void*)wp;
    wp += (bytes + 255) & ~(size_t)255;
    return p;
  };
  float*  Hbuf   = (float*)alloc(12L * 64 * IMG * 4);
  bf16_t* HBbf   = (bf16_t*)alloc(12L * 64 * IMG * 2);
  float*  FFTOUT = (float*)alloc(12L * 64 * IMG * 4);
  float*  RE1    = (float*)alloc(12L * 64 * HF * 4);
  float*  IM1    = (float*)alloc(12L * 64 * HF * 4);
  float*  RE2    = (float*)alloc(12L * 64 * HF * 4);
  float*  IM2    = (float*)alloc(12L * 64 * HF * 4);
  float*  MAGO   = (float*)alloc(12L * 64 * HF * 4);
  float*  PHAO   = (float*)alloc(12L * 64 * HF * 4);
  bf16_t* BBbf   = (bf16_t*)alloc(12L * 128 * HF * 2);
  float*  OFF18  = (float*)alloc(12L * 18 * IMG * 4);
  bf16_t* OFF3b  = (bf16_t*)alloc(12L * 3 * IMG * 2);
  bf16_t* DEF3b  = (bf16_t*)alloc(12L * 3 * IMG * 2);
  bf16_t* TMP64b = (bf16_t*)alloc(12L * 64 * IMG * 2);
  bf16_t* Qbf    = (bf16_t*)alloc(4L * 64 * IMG * 2);
  bf16_t* Kbf    = (bf16_t*)alloc(4L * 64 * IMG * 2);
  bf16_t* Vbf    = (bf16_t*)alloc(4L * 64 * IMG * 2);
  bf16_t* DWTb   = (bf16_t*)alloc(4L * 64 * IMG * 2);
  float*  Sf     = (float*)alloc(512L * 4096 * 4);
  bf16_t* Pbf    = (bf16_t*)alloc(512L * 4096 * 2);
  bf16_t* CAT1b  = (bf16_t*)alloc(4L * 128 * IMG * 2);
  bf16_t* CAT2b  = (bf16_t*)alloc(4L * 128 * IMG * 2);
  float*  HH1f   = (float*)alloc(4L * 64 * IMG * 4);
  bf16_t* HH1b   = (bf16_t*)alloc(4L * 64 * IMG * 2);
  bf16_t* P0b    = (bf16_t*)alloc(4L * 64 * IMG * 2);
  bf16_t* P1b    = (bf16_t*)alloc(4L * 64 * IMG * 2);
  float*  DBX    = (float*)alloc(12L * 64 * IMG * 4);
  float*  DBF    = (float*)alloc(12L * 64 * IMG * 4);
  float*  OUTX   = (float*)alloc(12L * 3 * IMG * 4);
  float*  OUTF   = (float*)alloc(12L * 3 * IMG * 4);
  // padded GEMM weights: 64 rows x Kp cols each
  bf16_t* choffpw_bf = (bf16_t*)alloc(64L * 32 * 2);
  bf16_t* chpw_bf    = (bf16_t*)alloc(64L * 32 * 2);
  bf16_t* ctoffpw_bf = (bf16_t*)alloc(64L * 64 * 2);
  bf16_t* ctpw_bf    = (bf16_t*)alloc(64L * 64 * 2);
  bf16_t* capw_bf    = (bf16_t*)alloc(6L * 64 * 64 * 2);
  bf16_t* fd1_bf     = (bf16_t*)alloc(2L * 64 * 64 * 2);
  bf16_t* fd2_bf     = (bf16_t*)alloc(2L * 64 * 96 * 2);
  bf16_t* fconv_bf   = (bf16_t*)alloc(2L * 64 * 128 * 2);
  // repacked conv weights [L][9][64][Ci]
  bf16_t* dbw_bf     = (bf16_t*)alloc(552960L * 2);
  bf16_t* convHH_bf  = (bf16_t*)alloc(73728L * 2);

  auto nb = [](long n) -> dim3 {
    long b = (n + 255) / 256;
    if (b > 262144) b = 262144;
    return dim3((unsigned)b);
  };
  auto padw = [&](const float* s, bf16_t* d, int M, int K, int Kp) {
    padw_kernel<<<nb(64L * Kp), 256, 0, stream>>>(s, d, M, K, Kp, 64L * Kp);
  };
  padw(ch_off_pw, choffpw_bf, 18, 3, 32);
  padw(ch_pw,     chpw_bf,    64, 3, 32);
  padw(ct_off_pw, ctoffpw_bf, 18, 64, 64);
  padw(ct_pw,     ctpw_bf,    3, 64, 64);
  for (int s = 0; s < 6; ++s) padw(ca_pw_w + s * 4096, capw_bf + s * 4096, 64, 64, 64);
  for (int i = 0; i < 2; ++i) padw(fd1_w + i * 2048, fd1_bf + i * 4096, 32, 64, 64);
  for (int i = 0; i < 2; ++i) padw(fd2_w + i * 3072, fd2_bf + i * 6144, 32, 96, 96);
  for (int i = 0; i < 2; ++i) padw(fconv_w + i * 8192, fconv_bf + i * 8192, 64, 128, 128);
  packconvw_kernel<<<nb(552960), 256, 0, stream>>>(db_w, dbw_bf, 64, 552960L);
  packconvw_kernel<<<nb(73728), 256, 0, stream>>>(convHH_w, convHH_bf, 128, 73728L);

  auto gemm = [&](const bf16_t* A, long a_so, long a_si, int lda,
                  const bf16_t* B, long b_so, long b_si, int ldb, int btrans,
                  float* C, bf16_t* Cb, long c_so, long c_si, int ldc, int binner,
                  int M, int N, int K, int Kp, int nbatch,
                  float alpha, const float* scale, const float* shift, int op, float slope) {
    wmma_gemm_kernel<<<dim3(N / 16, nbatch), 128, 0, stream>>>(
        A, a_so, a_si, lda, B, b_so, b_si, ldb, btrans,
        C, Cb, c_so, c_si, ldc, binner, M, N, K, Kp, alpha, scale, shift, op, slope);
  };
  auto conv3 = [&](const bf16_t* X, long xs, const bf16_t* W, const float* bias,
                   const float* res, long rs, float* Y, bf16_t* Yb, long ys,
                   int Cin, int dil, int leaky, int nimg) {
    wmma_conv3x3_kernel<<<dim3(IMG / 16, nimg), 128, 0, stream>>>(
        X, xs, W, bias, res, rs, Y, Yb, ys, Cin, dil, leaky, 0.01f);
  };

  const float bninv = 0.9999950000374997f;   // 1/sqrt(1+1e-5)

  // ============ Stage 1: patch_embed (ch) on x -> h ============
  dw3x3_kernel<<<nb(589824), 256, 0, stream>>>(x, ch_off_dw, nullptr, nullptr, OFF3b, 3, 589824L);
  gemm(choffpw_bf, 0, 0, 32, OFF3b, 3L * IMG, 0, IMG, 0,
       OFF18, nullptr, 18L * IMG, 0, IMG, 1, 18, IMG, 3, 32, 12, 1.f, nullptr, nullptr, 1, 0.f);
  deform_kernel<<<nb(589824), 256, 0, stream>>>(x, OFF18, ch_dcn_w, nullptr, DEF3b, 3, 589824L);
  gemm(chpw_bf, 0, 0, 32, DEF3b, 3L * IMG, 0, IMG, 0,
       Hbuf, HBbf, 64L * IMG, 0, IMG, 1, 64, IMG, 3, 32, 12, 1.f, nullptr, nullptr, 2, 0.f);

  // ============ Stage 2: FFT block ============
  dft_row_kernel<<<dim3(98304), 128, 0, stream>>>(Hbuf, RE1, IM1);
  dft_col_kernel<<<dim3(49920), 128, 0, stream>>>(RE1, IM1, RE2, IM2, -1.f, 1.f);
  magpha_kernel<<<nb(6389760), 256, 0, stream>>>(RE2, IM2, mag_ptr, pha_ptr, 6389760L);

  for (int br = 0; br < 2; ++br) {
    const float* src = br ? pha_ptr : mag_ptr;
    float* dst = br ? PHAO : MAGO;
    cvt_kernel<<<nb(6389760), 256, 0, stream>>>(src, 64L * HF, nullptr, BBbf, 128L * HF, 64L * HF, 12);
    gemm(fd1_bf + br * 4096, 0, 0, 64, BBbf, 128L * HF, 0, HF, 0,
         nullptr, BBbf + 64L * HF, 128L * HF, 0, HF, 1,
         32, HF, 64, 64, 12, bninv, fd1_g + br * 32, fd1_bt + br * 32, 3, 0.1f);
    gemm(fd2_bf + br * 6144, 0, 0, 96, BBbf, 128L * HF, 0, HF, 0,
         nullptr, BBbf + 96L * HF, 128L * HF, 0, HF, 1,
         32, HF, 96, 96, 12, bninv, fd2_g + br * 32, fd2_bt + br * 32, 3, 0.1f);
    gemm(fconv_bf + br * 8192, 0, 0, 128, BBbf, 128L * HF, 0, HF, 0,
         dst, nullptr, 64L * HF, 0, HF, 1, 64, HF, 128, 128, 12, 1.f, nullptr, nullptr, 0, 0.f);
  }
  polar_kernel<<<nb(6389760), 256, 0, stream>>>(MAGO, PHAO, RE2, IM2, 6389760L);
  dft_col_kernel<<<dim3(49920), 128, 0, stream>>>(RE2, IM2, RE1, IM1, 1.f, 1.f / 128.f);
  idft_row_add_kernel<<<dim3(98304), 128, 0, stream>>>(RE1, IM1, Hbuf, FFTOUT);

  // ============ Stage 3: cross attention ============
  const float* x_HL = Hbuf;
  const float* x_LH = Hbuf + 4L * 64 * IMG;
  const float* x_HH = Hbuf + 8L * 64 * IMG;
  auto depthq = [&](const float* src, int s, int j, bf16_t* dst) {
    dw3x3_kernel<<<nb(4L * 64 * IMG), 256, 0, stream>>>(
        src, ca_dw_w + (long)(s * 3 + j) * 64 * 9, ca_dw_b + (s * 3 + j) * 64,
        nullptr, DWTb, 64, 4L * 64 * IMG);
    gemm(capw_bf + (long)(s * 3 + j) * 4096, 0, 0, 64,
         DWTb, 64L * IMG, 0, IMG, 0,
         nullptr, dst, 64L * IMG, 0, IMG, 1,
         64, IMG, 64, 64, 4, 1.f, nullptr, ca_pw_b + (s * 3 + j) * 64, 0, 0.f);
  };
  auto attn_spo = [&](bf16_t* catdst) {
    gemm(Qbf, 64L * IMG, 128, IMG, Kbf, 64L * IMG, 128, IMG, 1,
         Sf, nullptr, 128L * 4096, 4096, 64, 128,
         64, 64, 128, 128, 512, 0.3535533906f, nullptr, nullptr, 0, 0.f);
    softmax64_kernel<<<dim3(128), 256, 0, stream>>>(Sf, Pbf, 32768);
    gemm(Pbf, 128L * 4096, 4096, 64, Vbf, 64L * IMG, 128, IMG, 0,
         nullptr, catdst, 128L * IMG, 128, IMG, 128,
         64, 128, 64, 64, 512, 1.f, nullptr, nullptr, 0, 0.f);
  };
  depthq(x_HH, 0, 1, Kbf);
  depthq(x_HH, 0, 2, Vbf);
  depthq(x_LH, 0, 0, Qbf);
  attn_spo(CAT1b);
  copyb_kernel<<<nb(4L * 64 * IMG), 256, 0, stream>>>(CAT1b, 128L * IMG, CAT2b, 128L * IMG, 64L * IMG, 4);
  depthq(x_HH, 1, 1, Kbf);
  depthq(x_HH, 1, 2, Vbf);
  depthq(x_HL, 1, 0, Qbf);
  attn_spo(CAT1b + 64L * IMG);
  depthq(FFTOUT, 1, 0, Qbf);
  attn_spo(CAT2b + 64L * IMG);

  // ============ Stage 4: convHH + dilated blocks ============
  auto dilated = [&](const bf16_t* inb, const float* inf, int set, float* outslice) {
    const int dl[5] = {1, 2, 3, 2, 1};
    const bf16_t* cur = inb;
    bf16_t* pp[2] = {P0b, P1b};
    for (int i = 0; i < 4; ++i) {
      conv3(cur, 64L * IMG, dbw_bf + (long)(set * 5 + i) * 9 * 64 * 64,
            db_b + (set * 5 + i) * 64, nullptr, 0,
            nullptr, pp[i & 1], 64L * IMG, 64, dl[i], 1, 4);
      cur = pp[i & 1];
    }
    conv3(cur, 64L * IMG, dbw_bf + (long)(set * 5 + 4) * 9 * 64 * 64,
          db_b + (set * 5 + 4) * 64, inf, 64L * IMG,
          outslice, nullptr, 64L * IMG, 64, dl[4], 0, 4);
  };
  conv3(CAT1b, 128L * IMG, convHH_bf, convHH_b, nullptr, 0, HH1f, HH1b, 64L * IMG, 128, 1, 0, 4);
  dilated(HH1b, HH1f, 2, DBX + 8L * 64 * IMG);
  dilated(HBbf, x_HL, 1, DBX);
  dilated(HBbf + 4L * 64 * IMG, x_LH, 0, DBX + 4L * 64 * IMG);
  conv3(CAT2b, 128L * IMG, convHH_bf, convHH_b, nullptr, 0, HH1f, HH1b, 64L * IMG, 128, 1, 0, 4);
  dilated(HH1b, HH1f, 2, DBF + 8L * 64 * IMG);
  cvt_kernel<<<nb(4L * 64 * IMG), 256, 0, stream>>>(FFTOUT, 0, DBF, nullptr, 0, 4L * 64 * IMG, 1);
  cvt_kernel<<<nb(4L * 64 * IMG), 256, 0, stream>>>(Hbuf + 4L * 64 * IMG, 0,
                                                    DBF + 4L * 64 * IMG, nullptr, 0, 4L * 64 * IMG, 1);

  // ============ Stage 5: patch_embed (ct) x2 ============
  auto pe_ct = [&](const float* in, float* out) {
    dw3x3_kernel<<<nb(12L * 64 * IMG), 256, 0, stream>>>(in, ct_off_dw, nullptr, nullptr, TMP64b, 64, 12L * 64 * IMG);
    gemm(ctoffpw_bf, 0, 0, 64, TMP64b, 64L * IMG, 0, IMG, 0,
         OFF18, nullptr, 18L * IMG, 0, IMG, 1, 18, IMG, 64, 64, 12, 1.f, nullptr, nullptr, 1, 0.f);
    deform_kernel<<<nb(12L * 64 * IMG), 256, 0, stream>>>(in, OFF18, ct_dcn_w, nullptr, TMP64b, 64, 12L * 64 * IMG);
    gemm(ctpw_bf, 0, 0, 64, TMP64b, 64L * IMG, 0, IMG, 0,
         out, nullptr, 3L * IMG, 0, IMG, 1, 3, IMG, 64, 64, 12, 1.f, nullptr, nullptr, 2, 0.f);
  };
  pe_ct(DBX, OUTX);
  pe_ct(DBF, OUTF);

  // ============ Stage 6: final conv + residual ============
  final_conv_kernel<<<nb(589824), 256, 0, stream>>>(OUTX, OUTF, conv_w, conv_b, x, out_ptr, 589824L);
}
